// LSTMKeywordSpotter_137438954423
// MI455X (gfx1250) — matrix operand order repaired and checked by reading, not compiled, not run on hardware
//
#include <hip/hip_runtime.h>
#include <hip/hip_bf16.h>

typedef __attribute__((ext_vector_type(16))) __bf16 v16bf;
typedef __attribute__((ext_vector_type(8)))  float  v8f;

union FragBF {
  v16bf v;
  uint4 q[2];
};

// CDNA5 async global->LDS copy (tracked by ASYNCcnt), ISA §10 / §15.18.3
__device__ __forceinline__ void async_ld_lds_b128(const void* gptr, void* lptr) {
  uint32_t lds_off = (uint32_t)(uintptr_t)lptr;   // LDS aperture: low 32 bits = LDS addr
  uint64_t gaddr   = (uint64_t)(uintptr_t)gptr;
  asm volatile("global_load_async_to_lds_b128 %0, %1, off"
               :: "v"(lds_off), "v"(gaddr) : "memory");
}
__device__ __forceinline__ void wait_async0() {
  asm volatile("s_wait_asynccnt 0x0" ::: "memory");
}

// exact piecewise fixed-point sigmoid from reference
__device__ __forceinline__ float sigq(float a) {
  float x = a * 256.0f;
  float m  = (x <= 768.0f && x >= -768.0f) ? 1.0f : 0.0f;
  float hi = (x > 768.0f) ? 256.0f : 0.0f;
  return (hi + floorf(43.0f * x * m * (1.0f / 256.0f)) + 128.0f * m) * (1.0f / 256.0f);
}
__device__ __forceinline__ float tanhq(float a) {
  return fminf(1.0f, fmaxf(-1.0f, a));
}

// ---------------------------------------------------------------------------
// Kernel 1: gather embedding rows per token, convert f32 -> bf16 (packed
// 2x bf16 per 32-bit store)
// ---------------------------------------------------------------------------
__global__ void gather_embed_kernel(const int* __restrict__ tokens,
                                    const float* __restrict__ emb,
                                    __bf16* __restrict__ X1) {
  int row = blockIdx.x;                       // row = n*T + t
  int tok = tokens[row];
  const float2* src = (const float2*)(emb + (size_t)tok * 400);
  unsigned int* dst = (unsigned int*)(X1 + (size_t)row * 400);
  for (int c = threadIdx.x; c < 200; c += blockDim.x) {
    float2 v = src[c];
    union { __bf16 h[2]; unsigned int u; } p;
    p.h[0] = (__bf16)v.x;
    p.h[1] = (__bf16)v.y;
    dst[c] = p.u;
  }
}

// ---------------------------------------------------------------------------
// Kernel 2: convert / pack weights to bf16 (pads zeroed)
// Wx1c: [320][416]   Wh1c: [320][96]   Wc2: [320][192] = [Wx2 | 0 | Wh2 | 0]
// ---------------------------------------------------------------------------
__global__ void convert_weights_kernel(const float* __restrict__ Wx1,
                                       const float* __restrict__ Wh1,
                                       const float* __restrict__ Wx2,
                                       const float* __restrict__ Wh2,
                                       __bf16* __restrict__ Wx1c,
                                       __bf16* __restrict__ Wh1c,
                                       __bf16* __restrict__ Wc2) {
  const int total = 320 * 416 + 320 * 96 + 320 * 192;
  for (int i = blockIdx.x * blockDim.x + threadIdx.x; i < total;
       i += gridDim.x * blockDim.x) {
    if (i < 320 * 416) {
      int r = i / 416, c = i % 416;
      Wx1c[i] = (__bf16)(c < 400 ? Wx1[r * 400 + c] : 0.0f);
    } else if (i < 320 * 416 + 320 * 96) {
      int j = i - 320 * 416;
      int r = j / 96, c = j % 96;
      Wh1c[j] = (__bf16)(c < 80 ? Wh1[r * 80 + c] : 0.0f);
    } else {
      int j = i - 320 * 416 - 320 * 96;
      int r = j / 192, c = j % 192;
      float v = 0.0f;
      if (c < 80)                  v = Wx2[r * 80 + c];
      else if (c >= 96 && c < 176) v = Wh2[r * 80 + (c - 96)];
      Wc2[j] = (__bf16)v;
    }
  }
}

// ---------------------------------------------------------------------------
// Kernel 3: big non-recurrent GEMM  G1 = X1 @ Wx1^T + b1
// B tile (320 x 32k, 20KB) staged into LDS double-buffered via async
// global->LDS loads; all 8 waves share it. A fragments pipelined 1 k-step
// ahead in registers; B fragments pipelined 2 tiles deep from LDS.
// C written as [t][n][320] f32.
// ---------------------------------------------------------------------------
__device__ __forceinline__ void gemm_loadA(const __bf16* arow, int ks, int half,
                                           FragBF& a) {
  a.q[0] = *(const uint4*)(arow + ks * 32 + half * 8);
  if (ks == 12)
    a.q[1] = make_uint4(0u, 0u, 0u, 0u);                 // K pad 400..415
  else
    a.q[1] = *(const uint4*)(arow + ks * 32 + 16 + half * 8);
}

__global__ void __launch_bounds__(256)
gemm_pre_kernel(const __bf16* __restrict__ A,
                const __bf16* __restrict__ B,
                const float* __restrict__ bias,
                float* __restrict__ C) {
  __shared__ __bf16 bstage[2][320][40];   // 40-stride spreads LDS banks; 50KB
  const int tid  = threadIdx.x;
  const int lane = tid & 31;
  const int wv   = tid >> 5;
  const int half = lane >> 4;
  const int nn   = lane & 15;
  const int mt   = blockIdx.x * 8 + wv;                  // M tile index
  const __bf16* arow = A + (size_t)(mt * 16 + nn) * 400;

  // ---- stage ks=0 into buffer 0 (async), preload A(0)
  for (int i = tid; i < 1280; i += 256) {
    int row = i >> 2, h16 = i & 3;
    async_ld_lds_b128(B + (size_t)row * 416 + 0 * 32 + h16 * 8,
                      &bstage[0][row][h16 * 8]);
  }
  FragBF aC;
  gemm_loadA(arow, 0, half, aC);

  v8f zero;
#pragma unroll
  for (int r = 0; r < 8; ++r) zero[r] = 0.0f;
  v8f acc[20];
#pragma unroll
  for (int n = 0; n < 20; ++n) acc[n] = zero;

  wait_async0();
  __syncthreads();

  for (int ks = 0; ks < 13; ++ks) {
    const int cur = ks & 1;
    // stage next k-step while computing this one
    if (ks < 12) {
      for (int i = tid; i < 1280; i += 256) {
        int row = i >> 2, h16 = i & 3;
        async_ld_lds_b128(B + (size_t)row * 416 + (ks + 1) * 32 + h16 * 8,
                          &bstage[cur ^ 1][row][h16 * 8]);
      }
    }
    FragBF aN;
    aN.q[0] = make_uint4(0u, 0u, 0u, 0u);
    aN.q[1] = make_uint4(0u, 0u, 0u, 0u);
    if (ks < 12) gemm_loadA(arow, ks + 1, half, aN);

    // depth-2 pipelined B-fragment loads from LDS
    FragBF bq[3];
#pragma unroll
    for (int p = 0; p < 2; ++p) {
      const __bf16* brow = &bstage[cur][p * 16 + nn][half * 8];
      bq[p].q[0] = *(const uint4*)(brow);
      bq[p].q[1] = *(const uint4*)(brow + 16);
    }
#pragma unroll
    for (int n = 0; n < 20; ++n) {
      if (n + 2 < 20) {
        const __bf16* brow = &bstage[cur][(n + 2) * 16 + nn][half * 8];
        bq[(n + 2) % 3].q[0] = *(const uint4*)(brow);
        bq[(n + 2) % 3].q[1] = *(const uint4*)(brow + 16);
      }
      acc[n] = __builtin_amdgcn_wmma_f32_16x16x32_bf16(
          false, aC.v, false, bq[n % 3].v, (short)0, acc[n], false, false);
    }
    wait_async0();
    __syncthreads();
    aC = aN;
  }

  const size_t rowbase = (size_t)mt * 16;
#pragma unroll
  for (int n = 0; n < 20; ++n) {
    float bv = bias[n * 16 + nn];
#pragma unroll
    for (int r = 0; r < 8; ++r) {
      size_t grow = rowbase + r + half * 8;              // = n_*1024 + t_
      size_t n_ = grow >> 10;
      size_t t_ = grow & 1023;
      C[(t_ * 128 + n_) * 320 + n * 16 + nn] = acc[n][r] + bv;
    }
  }
}

// ---------------------------------------------------------------------------
// Kernel 4: LSTM recurrence. One block = one 16-row batch tile (8 independent
// blocks). 5 waves; wave w owns hidden cols [16w,16w+16) of all four gates.
// Weight B-fragments stay in VGPRs for the whole t-loop. h (and staged x for
// layer 2) live in PING-PONG LDS buffers -> one barrier per step. The G1
// pre-activation tile and the x_t staging quad for step t+1 are loaded into
// registers during step t (software pipelining). All A-fragment DS loads are
// issued as one clause before the WMMA chain.
//   layer1: KSTEPS=3, pre = G1 (x-part precomputed), writes h1_all
//   layer2: KSTEPS=6 over [x_t | h], bias-only init, computes fc at t=T-1
// ---------------------------------------------------------------------------
template <int KSTEPS, bool HAS_PRE, bool HAS_X, bool DO_FC>
__global__ void __launch_bounds__(160)
lstm_rec_kernel(const float* __restrict__ pre,      // [1024][128][320] or null
                const __bf16* __restrict__ W,       // [320][KSTEPS*32]
                const float* __restrict__ bias,     // [320] when !HAS_PRE
                const __bf16* __restrict__ xsrc,    // h1_all [1024][128][80]
                __bf16* __restrict__ hout,          // h1_all or null
                const float* __restrict__ fcw,      // fc_w [11][80]
                const float* __restrict__ fcb,      // fc_b [11]
                float* __restrict__ out) {          // [128]
  __shared__ __bf16 lds[2][16][200];                // ping-pong; stride 200
  const int tid  = threadIdx.x;
  const int lane = tid & 31;
  const int wv   = tid >> 5;                        // 0..4
  const int half = lane >> 4;
  const int nn   = lane & 15;
  const int n0   = blockIdx.x * 16;                 // batch tile base
  constexpr int HCOL = HAS_X ? 96 : 0;

  // zero both buffers (establishes zero K-pads at cols 80..95 / 176..191)
  for (int i = tid; i < 2 * 16 * 200 / 8; i += 160)
    ((uint4*)&lds[0][0][0])[i] = make_uint4(0u, 0u, 0u, 0u);
  __syncthreads();

  // hoist weight B-fragments into registers (constant over time)
  FragBF bw[4][KSTEPS];
#pragma unroll
  for (int gg = 0; gg < 4; ++gg) {
    const __bf16* wr = W + (size_t)(gg * 80 + wv * 16 + nn) * (KSTEPS * 32);
#pragma unroll
    for (int ks = 0; ks < KSTEPS; ++ks) {
      bw[gg][ks].q[0] = *(const uint4*)(wr + ks * 32 + half * 16);
      bw[gg][ks].q[1] = *(const uint4*)(wr + ks * 32 + half * 16 + 8);
    }
  }
  float bg[4];
#pragma unroll
  for (int gg = 0; gg < 4; ++gg)
    bg[gg] = HAS_PRE ? 0.0f : bias[gg * 80 + wv * 16 + nn];

  float cst[8];
#pragma unroll
  for (int r = 0; r < 8; ++r) cst[r] = 0.0f;

  // software-pipelined inputs for the *current* step
  float npre[4][8];
  uint4 xreg = make_uint4(0u, 0u, 0u, 0u);
  const int rr = tid / 10, cc = tid % 10;           // x staging slot (160 uint4)

  if (HAS_PRE) {
    const float* pb = pre + (size_t)n0 * 320 + wv * 16 + nn;   // t = 0
#pragma unroll
    for (int gg = 0; gg < 4; ++gg)
#pragma unroll
      for (int r = 0; r < 8; ++r)
        npre[gg][r] = pb[(size_t)(r + half * 8) * 320 + gg * 80];
  }
  if (HAS_X)
    xreg = *(const uint4*)(xsrc + ((size_t)0 * 128 + n0 + rr) * 80 + cc * 8);

  for (int t = 0; t < 1024; ++t) {
    const int cur = t & 1;
    if (HAS_X)   // stage x_t (already in registers) into this step's buffer
      ((uint4*)&lds[cur][rr][0])[cc] = xreg;
    __syncthreads();   // h(t-1) writes + x(t) staging visible

    v8f acc[4];
    if (HAS_PRE) {
#pragma unroll
      for (int gg = 0; gg < 4; ++gg)
#pragma unroll
        for (int r = 0; r < 8; ++r)
          acc[gg][r] = npre[gg][r];
    } else {
#pragma unroll
      for (int gg = 0; gg < 4; ++gg)
#pragma unroll
        for (int r = 0; r < 8; ++r)
          acc[gg][r] = bg[gg];
    }

    // issue all A-fragment DS loads as one clause
    FragBF a[KSTEPS];
#pragma unroll
    for (int ks = 0; ks < KSTEPS; ++ks) {
      const __bf16* ar = &lds[cur][nn][ks * 32];
      a[ks].q[0] = *(const uint4*)(ar + half * 16);
      a[ks].q[1] = *(const uint4*)(ar + half * 16 + 8);
    }

    // kick off loads for step t+1 (latency hidden behind WMMA + activations)
    if (t + 1 < 1024) {
      if (HAS_PRE) {
        const float* pb = pre + ((size_t)(t + 1) * 128 + n0) * 320 + wv * 16 + nn;
#pragma unroll
        for (int gg = 0; gg < 4; ++gg)
#pragma unroll
          for (int r = 0; r < 8; ++r)
            npre[gg][r] = pb[(size_t)(r + half * 8) * 320 + gg * 80];
      }
      if (HAS_X)
        xreg = *(const uint4*)(xsrc + ((size_t)(t + 1) * 128 + n0 + rr) * 80 + cc * 8);
    }

#pragma unroll
    for (int ks = 0; ks < KSTEPS; ++ks)
#pragma unroll
      for (int gg = 0; gg < 4; ++gg)
        acc[gg] = __builtin_amdgcn_wmma_f32_16x16x32_bf16(
            false, a[ks].v, false, bw[gg][ks].v, (short)0, acc[gg], false, false);

#pragma unroll
    for (int r = 0; r < 8; ++r) {
      float iv = sigq(acc[0][r]);
      float fv = sigq(acc[1][r]);
      float gv = tanhq(acc[2][r]);
      float ov = sigq(acc[3][r]);
      float cn = fv * cst[r] + iv * gv;
      cst[r] = cn;
      float hv = ov * tanhq(cn);
      __bf16 hb = (__bf16)hv;
      // write h(t) into the *other* buffer; no conflict with ongoing reads
      lds[cur ^ 1][r + half * 8][HCOL + wv * 16 + nn] = hb;
      if (hout)
        hout[((size_t)t * 128 + n0 + r + half * 8) * 80 + wv * 16 + nn] = hb;
    }
  }

  if (DO_FC) {
    __syncthreads();
    // h(T-1) lives in buffer (T & 1) == 0
    if (wv == 0 && lane < 16) {
      float s = fcb[10];
      for (int k = 0; k < 80; ++k)
        s += (float)lds[0][lane][HCOL + k] * fcw[10 * 80 + k];
      out[n0 + lane] = sigq(s);
    }
  }
}

// ---------------------------------------------------------------------------
extern "C" void kernel_launch(void* const* d_in, const int* in_sizes, int n_in,
                              void* d_out, int out_size, void* d_ws,
                              size_t ws_size, hipStream_t stream) {
  (void)in_sizes; (void)n_in; (void)out_size; (void)ws_size;
  const int*   tokens = (const int*)d_in[0];
  const float* emb    = (const float*)d_in[1];
  const float* Wx1    = (const float*)d_in[2];
  const float* Wh1    = (const float*)d_in[3];
  const float* b1     = (const float*)d_in[4];
  const float* Wx2    = (const float*)d_in[5];
  const float* Wh2    = (const float*)d_in[6];
  const float* b2     = (const float*)d_in[7];
  const float* fcw    = (const float*)d_in[8];
  const float* fcb    = (const float*)d_in[9];
  float* out = (float*)d_out;

  char* ws = (char*)d_ws;
  size_t o = 0;
  auto carve = [&](size_t bytes) -> char* {
    char* p = ws + o;
    o += (bytes + 255) & ~(size_t)255;
    return p;
  };
  __bf16* X1    = (__bf16*)carve((size_t)131072 * 400 * 2);        // 105 MB
  float*  G1    = (float*) carve((size_t)1024 * 128 * 320 * 4);    // 168 MB
  __bf16* h1all = (__bf16*)carve((size_t)1024 * 128 * 80 * 2);     //  21 MB
  __bf16* Wx1c  = (__bf16*)carve((size_t)320 * 416 * 2);
  __bf16* Wh1c  = (__bf16*)carve((size_t)320 * 96 * 2);
  __bf16* Wc2   = (__bf16*)carve((size_t)320 * 192 * 2);

  gather_embed_kernel<<<131072, 256, 0, stream>>>(tokens, emb, X1);
  convert_weights_kernel<<<440, 512, 0, stream>>>(Wx1, Wh1, Wx2, Wh2,
                                                  Wx1c, Wh1c, Wc2);
  gemm_pre_kernel<<<1024, 256, 0, stream>>>(X1, Wx1c, b1, G1);
  lstm_rec_kernel<3, true, false, false><<<8, 160, 0, stream>>>(
      G1, Wh1c, nullptr, nullptr, h1all, nullptr, nullptr, nullptr);
  lstm_rec_kernel<6, false, true, true><<<8, 160, 0, stream>>>(
      nullptr, Wc2, b2, h1all, nullptr, fcw, fcb, out);
}
